// GraphConvolution_5875515261561
// MI455X (gfx1250) — matrix operand order repaired and checked by reading, not compiled, not run on hardware
//
#include <hip/hip_runtime.h>

#define N_NODES   50000
#define IN_F      512
#define OUT_F     512
#define NEG_SLOPE 0.01f

typedef __bf16 v16bf __attribute__((ext_vector_type(16)));
typedef float  v8f   __attribute__((ext_vector_type(8)));
typedef unsigned short u16;
typedef unsigned int   u32;

union V16Pack { v16bf v; uint4 q[2]; };

__device__ __forceinline__ u16 f32_to_bf16_rne(float f) {
    u32 u = __float_as_uint(f);
    u32 rounding = 0x7FFFu + ((u >> 16) & 1u);
    u += rounding;
    return (u16)(u >> 16);
}

__device__ __forceinline__ float lrelu(float v) {
    return v >= 0.f ? v : NEG_SLOPE * v;
}

// ---------------- conversion kernels (fp32 -> bf16, RNE) ----------------

__global__ __launch_bounds__(256) void convert_x_kernel(
    const float* __restrict__ x, const float* __restrict__ mask,
    u16* __restrict__ xb, int total4) {
    int i = blockIdx.x * blockDim.x + threadIdx.x;
    if (i >= total4) return;
    float4 xv = ((const float4*)x)[i];
    float4 mv = ((const float4*)mask)[i];
    uint2 p;
    p.x = (u32)f32_to_bf16_rne(xv.x * mv.x) | ((u32)f32_to_bf16_rne(xv.y * mv.y) << 16);
    p.y = (u32)f32_to_bf16_rne(xv.z * mv.z) | ((u32)f32_to_bf16_rne(xv.w * mv.w) << 16);
    ((uint2*)xb)[i] = p;
}

__global__ __launch_bounds__(256) void convert_w_kernel(
    const float* __restrict__ w, u16* __restrict__ wb, int total4) {
    int i = blockIdx.x * blockDim.x + threadIdx.x;
    if (i >= total4) return;
    float4 wv = ((const float4*)w)[i];
    uint2 p;
    p.x = (u32)f32_to_bf16_rne(wv.x) | ((u32)f32_to_bf16_rne(wv.y) << 16);
    p.y = (u32)f32_to_bf16_rne(wv.z) | ((u32)f32_to_bf16_rne(wv.w) << 16);
    ((uint2*)wb)[i] = p;
}

// ---------------- WMMA GEMM: h = leaky_relu((x*mask) @ W^T) ----------------
// Block: 256 threads = 8 waves. blockIdx.x = M-tile (16 rows), blockIdx.y = N-group.
// Each wave computes one 16x16 output tile; A strip (16 x 512 bf16) staged in LDS.

__global__ __launch_bounds__(256) void gemm_bf16_wmma_kernel(
    const u16* __restrict__ xb,   // [N_NODES, IN_F] bf16
    const u16* __restrict__ wb,   // [OUT_F, IN_F]   bf16
    float* __restrict__ h) {      // [N_NODES, OUT_F] f32
    __shared__ __align__(16) u16 lds_a[16 * IN_F];   // 16 KB

    const int mtile = blockIdx.x;           // 0..3124
    const int ngrp  = blockIdx.y;           // 0..3
    const int tid   = threadIdx.x;

    // Cooperative load of A strip: 16*512 u16 = 1024 uint4
    {
        const uint4* src = (const uint4*)(xb + (size_t)mtile * 16 * IN_F);
        uint4* dst = (uint4*)lds_a;
        #pragma unroll
        for (int i = 0; i < 4; ++i)
            dst[tid + i * 256] = src[tid + i * 256];
    }
    __syncthreads();

    const int wave  = tid >> 5;
    const int lane  = tid & 31;
    const int ntile = ngrp * 8 + wave;      // 0..31
    const int ln    = lane & 15;            // A: row M; B: col N; C: col N
    const int hi    = lane >> 4;            // lane half selector

    v8f c = {};                             // f32 accumulator, zero

    // B row base: B[k][n] = W[n][k]; this lane owns column n = ln
    const u16* brow = wb + (size_t)(ntile * 16 + ln) * IN_F;
    const u16* arow = &lds_a[ln * IN_F];

    #pragma unroll 4
    for (int k0 = 0; k0 < IN_F; k0 += 32) {
        // A 16x32 bf16 layout: lanes 0-15 hold K=[k0..k0+15] of row ln;
        // lanes 16-31 hold K=[k0+16..k0+31].
        V16Pack a;
        a.q[0] = *(const uint4*)(arow + k0 + hi * 16);
        a.q[1] = *(const uint4*)(arow + k0 + hi * 16 + 8);
        // B 32x16 bf16 layout: lane half 0 holds K=[k0..k0+15], half 1 K=[k0+16..k0+31]
        V16Pack b;
        const uint4* bp = (const uint4*)(brow + k0 + hi * 16);
        b.q[0] = bp[0];
        b.q[1] = bp[1];
        c = __builtin_amdgcn_wmma_f32_16x16x32_bf16(
            /*neg_a=*/false, a.v, /*neg_b=*/false, b.v,
            /*c_mod=*/(short)0, c, /*reuse_a=*/false, /*reuse_b=*/false);
    }

    // C/D layout: VGPR j -> row (j + 8*hi), col ln. Fuse leaky_relu.
    float* out_tile = h + (size_t)(mtile * 16) * OUT_F + ntile * 16;
    #pragma unroll
    for (int j = 0; j < 8; ++j) {
        out_tile[(size_t)(j + hi * 8) * OUT_F + ln] = lrelu(c[j]);
    }
}

// ---------------- edge scatter: out[row] += val * h[col] ----------------
// One wave (32 lanes) per edge; 512 floats = 128 float4, 4 float4 per lane.

__global__ __launch_bounds__(256) void scatter_kernel(
    const float* __restrict__ h, const float* __restrict__ vals,
    const int* __restrict__ rows, const int* __restrict__ cols,
    float* __restrict__ out, int n_edges) {
    int edge = blockIdx.x * (blockDim.x >> 5) + (threadIdx.x >> 5);
    int lane = threadIdx.x & 31;
    if (edge >= n_edges) return;
    const int   r = rows[edge];
    const int   col = cols[edge];
    const float v = vals[edge];
    const float4* hc = (const float4*)(h + (size_t)col * OUT_F);
    float* o = out + (size_t)r * OUT_F;
    #pragma unroll
    for (int i = 0; i < 4; ++i) {
        int q = lane + i * 32;
        float4 hv = hc[q];
        int base = q * 4;
        atomicAdd(o + base + 0, v * hv.x);
        atomicAdd(o + base + 1, v * hv.y);
        atomicAdd(o + base + 2, v * hv.z);
        atomicAdd(o + base + 3, v * hv.w);
    }
}

// ---------------- final elementwise leaky_relu ----------------

__global__ __launch_bounds__(256) void leaky_kernel(float* __restrict__ out, int total4) {
    int i = blockIdx.x * blockDim.x + threadIdx.x;
    if (i >= total4) return;
    float4 v = ((float4*)out)[i];
    v.x = lrelu(v.x); v.y = lrelu(v.y); v.z = lrelu(v.z); v.w = lrelu(v.w);
    ((float4*)out)[i] = v;
}

// ---------------- host launcher ----------------

extern "C" void kernel_launch(void* const* d_in, const int* in_sizes, int n_in,
                              void* d_out, int out_size, void* d_ws, size_t ws_size,
                              hipStream_t stream) {
    const float* x    = (const float*)d_in[0];   // [N_NODES, IN_F]
    const float* W    = (const float*)d_in[1];   // [OUT_F, IN_F]
    const float* mask = (const float*)d_in[2];   // [N_NODES, IN_F]
    const float* vals = (const float*)d_in[3];   // [E]
    const int*   rows = (const int*)d_in[4];     // [E]
    const int*   cols = (const int*)d_in[5];     // [E]
    float* out = (float*)d_out;

    const int n_edges = in_sizes[3];
    const int n_nodes = in_sizes[0] / IN_F;      // 50000 (multiple of 16)

    // Workspace layout: xb (bf16) | wb (bf16) | h (f32)
    const size_t XB_BYTES = (size_t)n_nodes * IN_F * sizeof(u16);     // 51.2 MB
    const size_t WB_BYTES = (size_t)OUT_F * IN_F * sizeof(u16);       // 0.5 MB
    char* ws = (char*)d_ws;
    u16*   xb = (u16*)ws;
    u16*   wb = (u16*)(ws + XB_BYTES);
    float* h  = (float*)(ws + XB_BYTES + WB_BYTES);

    // Zero the output accumulator (graph-capture-legal memset node).
    hipMemsetAsync(d_out, 0, (size_t)out_size * sizeof(float), stream);

    const int wtot4 = (OUT_F * IN_F) / 4;
    convert_w_kernel<<<(wtot4 + 255) / 256, 256, 0, stream>>>(W, wb, wtot4);

    const int xtot4 = (n_nodes * IN_F) / 4;
    convert_x_kernel<<<(xtot4 + 255) / 256, 256, 0, stream>>>(x, mask, xb, xtot4);

    dim3 ggrid(n_nodes / 16, OUT_F / 128);       // 3125 x 4
    gemm_bf16_wmma_kernel<<<ggrid, 256, 0, stream>>>(xb, wb, h);

    scatter_kernel<<<(n_edges + 7) / 8, 256, 0, stream>>>(h, vals, rows, cols, out, n_edges);

    const int otot4 = out_size / 4;
    leaky_kernel<<<(otot4 + 255) / 256, 256, 0, stream>>>(out, otot4);
}
